// MultiHeadAttentionWindow_77524159693115
// MI455X (gfx1250) — compile-verified, matrix operand order checked by reading, not run on hardware
//
#include <hip/hip_runtime.h>
#include <hip/hip_bf16.h>

#define KSEQ   4000
#define KP     4100
#define DMODEL 256
#define NH     8
#define DH     32
#define WINSZ  200
#define PADW   50
#define STEPW  100
#define NWIN   40
#define BATCH  8

typedef __attribute__((ext_vector_type(16))) _Float16 v16h;
typedef __attribute__((ext_vector_type(8)))  _Float16 v8h;
typedef __attribute__((ext_vector_type(8)))  float    v8f;

union V16U { v16h v; v8h h[2]; };

// A/B fragment loader for v_wmma_f32_16x16x32_f16.
// A (16x32, MxK): lane m = lane%16 owns row m; lanes 0-15 hold K 0..7 & 16..23,
// lanes 16-31 hold K 8..15 & 24..31 (two contiguous 8-half chunks per lane).
// B (32x16, KxN): symmetric — lane owns column n = lane%16 with identical K
// striping, so a row of the *transposed* B matrix loads the same way.
__device__ __forceinline__ v16h load_frag(const _Float16* rowp, int kbase, int hsel) {
    V16U u;
    u.h[0] = *(const v8h*)(rowp + kbase + hsel * 8);
    u.h[1] = *(const v8h*)(rowp + kbase + 16 + hsel * 8);
    return u.v;
}

__device__ __forceinline__ v8f wmma_f16(v16h a, v16h b, v8f c) {
    return __builtin_amdgcn_wmma_f32_16x16x32_f16(false, a, false, b, (short)0, c, false, false);
}

// Async cache->LDS DMA staging (ASYNCcnt path, bypasses VGPRs).
// LDS byte offset = low 32 bits of the generic pointer (LDS aperture rule).
__device__ __forceinline__ void async_copy_b128(unsigned lds_off, const void* gsrc) {
    unsigned long long src = (unsigned long long)(uintptr_t)gsrc;
    asm volatile("global_load_async_to_lds_b128 %0, %1, off"
                 :: "v"(lds_off), "v"(src) : "memory");
}
__device__ __forceinline__ void async_copy_b32(unsigned lds_off, const void* gsrc) {
    unsigned long long src = (unsigned long long)(uintptr_t)gsrc;
    asm volatile("global_load_async_to_lds_b32 %0, %1, off"
                 :: "v"(lds_off), "v"(src) : "memory");
}
__device__ __forceinline__ void async_wait0() {
    asm volatile("s_wait_asynccnt 0" ::: "memory");
}
__device__ __forceinline__ unsigned lds_off_of(const void* p) {
    return (unsigned)(uintptr_t)p;
}

// ---------------------------------------------------------------------------
// Kernel 1: convert+transpose the four weight matrices to f16: wt[mat][n][k]
// ---------------------------------------------------------------------------
__global__ __launch_bounds__(256) void mha_wconv_kernel(
    const float* __restrict__ wq, const float* __restrict__ wk,
    const float* __restrict__ wv, const float* __restrict__ wo,
    _Float16* __restrict__ wt)
{
    int idx = blockIdx.x * 256 + threadIdx.x;      // 4*65536 total, exact grid
    int mat = idx >> 16;
    int rem = idx & 65535;
    int n = rem >> 8, k = rem & 255;
    const float* src = (mat == 0) ? wq : (mat == 1) ? wk : (mat == 2) ? wv : wo;
    wt[idx] = (_Float16)src[k * DMODEL + n];       // wt[mat][n][k] = W[k][n]
}

// ---------------------------------------------------------------------------
// Kernel 2: fused Q/K/V projection over the edge-padded sequence.
// Q,K -> [b][h][row][32] f16 ; V -> transposed [b][h][32][row] f16
// ---------------------------------------------------------------------------
__global__ __launch_bounds__(256) void mha_proj_kernel(
    const float* __restrict__ xq, const float* __restrict__ xk,
    const float* __restrict__ xv, const _Float16* __restrict__ wt,
    const float* __restrict__ bq, const float* __restrict__ bk,
    const float* __restrict__ bv,
    _Float16* __restrict__ qh, _Float16* __restrict__ kh,
    _Float16* __restrict__ vt)
{
    __shared__ __align__(16) _Float16 sX[16 * DMODEL];   // 8 KB
    int rt = blockIdx.x;                 // row tile (padded rows), 0..256
    int b  = blockIdx.y;
    int tid = threadIdx.x, lane = tid & 31, wave = tid >> 5;
    int hsel = lane >> 4, nn = lane & 15;
    int rowbase = rt * 16;

    const float* srcs[3] = { xq, xk, xv };
    const float* bias[3] = { bq, bk, bv };

    for (int mat = 0; mat < 3; ++mat) {
        __syncthreads();
        // stage 16 padded rows of the source, f32 -> f16 (edge replicate pad)
        for (int e = tid; e < 16 * DMODEL; e += 256) {
            int r = e >> 8, c = e & 255;
            int src = rowbase + r - PADW;
            src = src < 0 ? 0 : (src > KSEQ - 1 ? KSEQ - 1 : src);
            sX[e] = (_Float16)srcs[mat][(b * KSEQ + src) * DMODEL + c];
        }
        __syncthreads();

        const _Float16* arow = sX + nn * DMODEL;     // A row m = lane%16
        for (int c2 = 0; c2 < 2; ++c2) {
            int ct  = wave * 2 + c2;                 // 16 column tiles / 8 waves
            int col = ct * 16 + nn;
            float b0 = bias[mat][col];
            v8f acc;
            #pragma unroll
            for (int i = 0; i < 8; ++i) acc[i] = b0;
            const _Float16* brow = wt + (mat * DMODEL + col) * DMODEL;
            __builtin_prefetch(brow, 0, 1);          // global_prefetch_b8
            #pragma unroll
            for (int kc = 0; kc < 8; ++kc) {         // K = 256 = 8 x 32
                v16h a = load_frag(arow, kc * 32, hsel);
                v16h w = load_frag(brow, kc * 32, hsel);
                acc = wmma_f16(a, w, acc);
            }
            int h = col >> 5, dw = col & 31;
            #pragma unroll
            for (int r = 0; r < 8; ++r) {
                int pr = rowbase + r + 8 * hsel;     // D row
                if (pr < KP) {
                    _Float16 v = (_Float16)acc[r];
                    if (mat == 0)      qh[((b * NH + h) * KP + pr) * DH + dw] = v;
                    else if (mat == 1) kh[((b * NH + h) * KP + pr) * DH + dw] = v;
                    else               vt[((b * NH + h) * DH + dw) * KP + pr] = v;
                }
            }
        }
    }
}

// ---------------------------------------------------------------------------
// Kernel 3: windowed attention. Block = (win, h, b), 8 waves.
// Waves 0..6 own row tiles rt=3..9 (rows 48..159; only 50..149 are emitted).
// Causal mask kills j>=160, so K/V windows are 160 rows.
// ---------------------------------------------------------------------------
__global__ __launch_bounds__(256) void mha_attn_kernel(
    const _Float16* __restrict__ qh, const _Float16* __restrict__ kh,
    const _Float16* __restrict__ vt, _Float16* __restrict__ ctx)
{
    __shared__ __align__(16) _Float16 sQ[112 * 32];      //  7168 B
    __shared__ __align__(16) _Float16 sK[160 * 32];      // 10240 B
    __shared__ __align__(16) _Float16 sV[32 * 160];      // 10240 B (V^T)
    __shared__ __align__(16) _Float16 sP[7 * 16 * 160];  // 35840 B
    int win = blockIdx.x, h = blockIdx.y, b = blockIdx.z;
    int tid = threadIdx.x, lane = tid & 31, wave = tid >> 5;
    int hsel = lane >> 4, nn = lane & 15;
    int base = b * NH + h;

    // ---- async-DMA staging (GLOBAL_LOAD_ASYNC_TO_LDS, ASYNCcnt path) ----
    {
        const _Float16* gk = kh + (base * KP + win * STEPW) * DH;      // 160x32, contiguous
        unsigned kb = lds_off_of(sK);
        for (int e = tid; e < (160 * 32) / 8; e += 256)
            async_copy_b128(kb + (unsigned)e * 16u, gk + e * 8);
        const _Float16* gq = qh + (base * KP + win * STEPW + 48) * DH; // 112x32, contiguous
        unsigned qb = lds_off_of(sQ);
        for (int e = tid; e < (112 * 32) / 8; e += 256)
            async_copy_b128(qb + (unsigned)e * 16u, gq + e * 8);
        const _Float16* gv = vt + base * DH * KP + win * STEPW;        // 32 strided rows
        unsigned vb = lds_off_of(sV);
        for (int e = tid; e < (32 * 160) / 2; e += 256) {
            int d = e / 80, j2 = (e % 80) * 2;                         // 4B granules
            async_copy_b32(vb + (unsigned)(d * 160 + j2) * 2u, gv + d * KP + j2);
        }
        async_wait0();
    }
    __syncthreads();

    int rt  = wave + 3;          // row tile index (valid for wave<7)
    int rtb = rt * 16;
    int nts = rt + 1;            // live S column tiles (lower triangle)
    v8f accS[10];
    float sm[8];

    if (wave < 7) {
        // ---- S = Q K^T : one 16x16x32 WMMA per col tile (d_h = 32 exactly)
        const _Float16* arow = sQ + (rtb - 48 + nn) * 32;
        v16h aq = load_frag(arow, 0, hsel);
        #pragma unroll
        for (int t = 0; t < 10; ++t) if (t < nts) {
            v16h bk_ = load_frag(sK + (t * 16 + nn) * 32, 0, hsel);
            v8f z = {0.f,0.f,0.f,0.f,0.f,0.f,0.f,0.f};
            accS[t] = wmma_f16(aq, bk_, z);
        }
        // ---- mask (scale * cauchy, zero diag, causal) + softmax in registers
        const float scale = 0.07071067811865475f;    // 1/sqrt(200)
        #pragma unroll
        for (int r = 0; r < 8; ++r) {
            int i = rtb + r + 8 * hsel;              // query row
            float mx = -1e30f;
            #pragma unroll
            for (int t = 0; t < 10; ++t) if (t < nts) {
                int j = t * 16 + nn;                 // key column
                float s;
                if (j > i)       s = -1e30f;
                else if (j == i) s = 0.f;            // cauchy diag = 0
                else {
                    float d = (float)(i - j);
                    s = accS[t][r] * scale / (1.f + 0.25f * d * d);
                }
                accS[t][r] = s;
                mx = fmaxf(mx, s);
            }
            #pragma unroll
            for (int m = 1; m < 16; m <<= 1) mx = fmaxf(mx, __shfl_xor(mx, m, 32));
            float sum = 0.f;
            #pragma unroll
            for (int t = 0; t < 10; ++t) if (t < nts) {
                float e = __expf(accS[t][r] - mx);
                accS[t][r] = e;
                sum += e;
            }
            #pragma unroll
            for (int m = 1; m < 16; m <<= 1) sum += __shfl_xor(sum, m, 32);
            sm[r] = sum;
        }
        // ---- write normalized P (f16) to this wave's LDS strip, zero-padded
        _Float16* pws = sP + wave * (16 * 160);
        uint4 z4; z4.x = z4.y = z4.z = z4.w = 0u;
        for (int e = lane; e < (16 * 160) / 8; e += 32) ((uint4*)pws)[e] = z4;
        #pragma unroll
        for (int t = 0; t < 10; ++t) if (t < nts) {
            #pragma unroll
            for (int r = 0; r < 8; ++r) {
                int m = r + 8 * hsel;
                pws[m * 160 + t * 16 + nn] = (_Float16)(accS[t][r] / sm[r]);
            }
        }
    }
    __syncthreads();             // all 8 waves participate

    if (wave < 7) {
        // ---- O = P V : K chunks of 32 over the live columns
        int nch = (rt + 2) >> 1;
        v8f o0 = {0.f,0.f,0.f,0.f,0.f,0.f,0.f,0.f};
        v8f o1 = {0.f,0.f,0.f,0.f,0.f,0.f,0.f,0.f};
        const _Float16* prow = sP + wave * (16 * 160) + nn * 160;
        #pragma unroll
        for (int c = 0; c < 5; ++c) if (c < nch) {
            v16h ap = load_frag(prow, c * 32, hsel);
            v16h b0 = load_frag(sV + nn * 160, c * 32, hsel);           // d = 0..15
            o0 = wmma_f16(ap, b0, o0);
            v16h b1 = load_frag(sV + (16 + nn) * 160, c * 32, hsel);    // d = 16..31
            o1 = wmma_f16(ap, b1, o1);
        }
        #pragma unroll
        for (int r = 0; r < 8; ++r) {
            int i = rtb + r + 8 * hsel;
            if (i >= PADW && i < WINSZ - PADW) {
                int p = win * STEPW + (i - PADW);
                int o = (b * KSEQ + p) * DMODEL + h * DH;
                ctx[o + nn]      = (_Float16)o0[r];
                ctx[o + 16 + nn] = (_Float16)o1[r];
            }
        }
    }
}

// ---------------------------------------------------------------------------
// Kernel 4: output projection  out = ctx @ Wo + bo  (f32 result)
// ---------------------------------------------------------------------------
__global__ __launch_bounds__(256) void mha_oproj_kernel(
    const _Float16* __restrict__ ctx, const _Float16* __restrict__ wt,
    const float* __restrict__ bo, float* __restrict__ out)
{
    __shared__ __align__(16) _Float16 sX[16 * DMODEL];
    int rt = blockIdx.x;                       // 0..1999 (32000 rows / 16)
    int tid = threadIdx.x, lane = tid & 31, wave = tid >> 5;
    int hsel = lane >> 4, nn = lane & 15;
    int rowbase = rt * 16;

    // async-DMA stage the 16x256 f16 ctx tile (8 KB contiguous)
    {
        const _Float16* gx = ctx + rowbase * DMODEL;
        unsigned xb = lds_off_of(sX);
        for (int e = tid; e < (16 * DMODEL) / 8; e += 256)
            async_copy_b128(xb + (unsigned)e * 16u, gx + e * 8);
        async_wait0();
    }
    __syncthreads();

    const _Float16* arow = sX + nn * DMODEL;
    const _Float16* wo = wt + 3 * DMODEL * DMODEL;    // Wo^T block
    for (int c2 = 0; c2 < 2; ++c2) {
        int ct = wave * 2 + c2;
        int col = ct * 16 + nn;
        float b0 = bo[col];
        v8f acc;
        #pragma unroll
        for (int i = 0; i < 8; ++i) acc[i] = b0;
        const _Float16* brow = wo + col * DMODEL;
        __builtin_prefetch(brow, 0, 1);               // global_prefetch_b8
        #pragma unroll
        for (int kc = 0; kc < 8; ++kc) {
            v16h a = load_frag(arow, kc * 32, hsel);
            v16h w = load_frag(brow, kc * 32, hsel);
            acc = wmma_f16(a, w, acc);
        }
        #pragma unroll
        for (int r = 0; r < 8; ++r) {
            int m = r + 8 * hsel;
            out[(rowbase + m) * DMODEL + col] = acc[r];
        }
    }
}

// ---------------------------------------------------------------------------
extern "C" void kernel_launch(void* const* d_in, const int* in_sizes, int n_in,
                              void* d_out, int out_size, void* d_ws, size_t ws_size,
                              hipStream_t stream) {
    const float* q  = (const float*)d_in[0];
    const float* k  = (const float*)d_in[1];
    const float* v  = (const float*)d_in[2];
    const float* Wq = (const float*)d_in[3];
    const float* bq = (const float*)d_in[4];
    const float* Wk = (const float*)d_in[5];
    const float* bk = (const float*)d_in[6];
    const float* Wv = (const float*)d_in[7];
    const float* bv = (const float*)d_in[8];
    const float* Wo = (const float*)d_in[9];
    const float* bo = (const float*)d_in[10];
    float* out = (float*)d_out;

    // workspace carve-up (bytes), all 16B-aligned
    char* ws = (char*)d_ws;
    const size_t WT_B  = (size_t)4 * DMODEL * DMODEL * 2;          //   512 KB
    const size_t QH_B  = (size_t)BATCH * NH * KP * DH * 2;         // ~16.8 MB
    _Float16* wt  = (_Float16*)(ws);
    _Float16* qh  = (_Float16*)(ws + WT_B);
    _Float16* kh  = (_Float16*)(ws + WT_B + QH_B);
    _Float16* vtb = (_Float16*)(ws + WT_B + 2 * QH_B);
    _Float16* ctx = (_Float16*)(ws + WT_B + 3 * QH_B);             // 16.4 MB

    mha_wconv_kernel<<<(4 * DMODEL * DMODEL) / 256, 256, 0, stream>>>(
        Wq, Wk, Wv, Wo, wt);

    mha_proj_kernel<<<dim3((KP + 15) / 16, BATCH), 256, 0, stream>>>(
        q, k, v, wt, bq, bk, bv, qh, kh, vtb);

    mha_attn_kernel<<<dim3(NWIN, NH, BATCH), 256, 0, stream>>>(
        qh, kh, vtb, ctx);

    mha_oproj_kernel<<<(BATCH * KSEQ) / 16, 256, 0, stream>>>(
        ctx, wt, bo, out);
}
